// TernaryWrapper_68178310856944
// MI455X (gfx1250) — matrix operand order repaired and checked by reading, not compiled, not run on hardware
//
#include <hip/hip_runtime.h>

typedef __attribute__((ext_vector_type(16))) __bf16 v16bf;
typedef __attribute__((ext_vector_type(8)))  __bf16 v8bf;
typedef __attribute__((ext_vector_type(8)))  float  v8f;
typedef __attribute__((ext_vector_type(4)))  float  v4f;

static constexpr int K_DIM = 1024;
static constexpr int N_DIM = 1024;

// ---------------------------------------------------------------------------
// Pass 1a: per-block partial sums of |W|  (1024 blocks x 1024 elements each)
// Fixed-tree reduction => bitwise deterministic across replays.
// ---------------------------------------------------------------------------
__global__ __launch_bounds__(256) void absmean_partial_kernel(
    const float* __restrict__ w, float* __restrict__ part) {
  __shared__ float smem[256];
  const int tid = threadIdx.x;
  const size_t base = (size_t)blockIdx.x * 1024 + tid;
  float v = fabsf(w[base]) + fabsf(w[base + 256]) +
            fabsf(w[base + 512]) + fabsf(w[base + 768]);
  smem[tid] = v;
  __syncthreads();
  for (int off = 128; off > 0; off >>= 1) {
    if (tid < off) smem[tid] += smem[tid + off];
    __syncthreads();
  }
  if (tid == 0) part[blockIdx.x] = smem[0];
}

// ---------------------------------------------------------------------------
// Pass 1b: reduce 1024 partials -> scale = mean(|W|)
// ---------------------------------------------------------------------------
__global__ __launch_bounds__(256) void absmean_final_kernel(
    const float* __restrict__ part, float* __restrict__ scale) {
  __shared__ float smem[256];
  const int tid = threadIdx.x;
  float v = part[tid] + part[tid + 256] + part[tid + 512] + part[tid + 768];
  smem[tid] = v;
  __syncthreads();
  for (int off = 128; off > 0; off >>= 1) {
    if (tid < off) smem[tid] += smem[tid + off];
    __syncthreads();
  }
  if (tid == 0) scale[0] = smem[0] * (1.0f / (1024.0f * 1024.0f));
}

// ---------------------------------------------------------------------------
// Pass 2: ternary-quantize W -> bf16 {-1,0,1}  (exact in bf16)
// q = clip(rint(w / max(scale, eps)), -1, 1)     (rintf == RNE == jnp.round)
// ---------------------------------------------------------------------------
__global__ __launch_bounds__(256) void quantize_kernel(
    const float* __restrict__ w, const float* __restrict__ scale_p,
    __bf16* __restrict__ tq) {
  const float inv = 1.0f / fmaxf(scale_p[0], 1e-8f);
  const int i = blockIdx.x * 256 + threadIdx.x;
  float q = rintf(w[i] * inv);
  q = fminf(1.0f, fmaxf(-1.0f, q));
  tq[i] = (__bf16)q;
}

// ---------------------------------------------------------------------------
// Pass 3: GEMM  out[M,N] = (x[M,K](bf16) @ T[N,K]^T) * scale + bias
// Block: 256 threads = 8 waves (2 m-waves x 4 n-waves) -> 128x256 block tile.
// Wave: 64x64 tile = 4x4 WMMA accumulators (128 acc VGPRs).
//   Per K-step(32): 8 B-loads + 16 A-loads + 32 cvt_pk + 16 WMMAs.
// A frag (16-bit 16x32): lane l: row = l&15, kb = (l>>4)*8;
//   VGPR0..3 = K kb..kb+7, VGPR4..7 = K kb+16..kb+23.
// B frag (32x16): lane l: col = l&15, kb = (l>>4)*8; elems 0..7 = K kb..kb+7, 8..15 = K kb+16..kb+23.
// C frag f32 16x16: VGPR r: lanes 0-15 -> (M=r, N=lane); 16-31 -> (M=r+8).
// Grid: x = N-tiles (fastest) so same-M blocks co-run and share x via L2.
// ---------------------------------------------------------------------------
__global__ __launch_bounds__(256) void ternary_gemm_kernel(
    const float* __restrict__ x, const __bf16* __restrict__ tq,
    const float* __restrict__ bias, const float* __restrict__ scale_p,
    float* __restrict__ out, int M) {
  const int lane = threadIdx.x & 31;
  const int wave = threadIdx.x >> 5;
  const int wm   = wave >> 2;            // 0..1  -> 64-row slab
  const int wn   = wave & 3;             // 0..3  -> 64-col slab
  const int m0   = blockIdx.y * 128 + wm * 64;
  const int n0   = blockIdx.x * 256 + wn * 64;
  if (m0 >= M) return;

  const int r  = lane & 15;
  const int hi = lane >> 4;

  const float*  xrow[4];
  const __bf16* trow[4];
#pragma unroll
  for (int i = 0; i < 4; ++i)
    xrow[i] = x + (size_t)(m0 + i * 16 + r) * K_DIM + hi * 8;
#pragma unroll
  for (int j = 0; j < 4; ++j)
    trow[j] = tq + (size_t)(n0 + j * 16 + r) * K_DIM + hi * 8;

  v8f acc[4][4] = {};

  for (int k = 0; k < K_DIM; k += 32) {
    // --- B fragments: two 16-byte chunks (K kb..kb+7, kb+16..kb+23) of bf16 ternary weights ---
    v16bf bfrag[4];
#pragma unroll
    for (int j = 0; j < 4; ++j) {
      v8bf blo = *(const v8bf*)(trow[j] + k);
      v8bf bhi = *(const v8bf*)(trow[j] + k + 16);
      bfrag[j] = __builtin_shufflevector(blo, bhi, 0, 1, 2, 3, 4, 5, 6, 7, 8,
                                         9, 10, 11, 12, 13, 14, 15);
    }
    // --- A fragments: fp32 x loaded 128b at a time, cvt -> bf16 in-reg,
    //     each frag immediately feeds 4 WMMAs (64 KFLOP per frag) ---
#pragma unroll
    for (int i = 0; i < 4; ++i) {
      const float* ap = xrow[i] + k;
      v4f f0 = *(const v4f*)(ap);
      v4f f1 = *(const v4f*)(ap + 4);
      v4f f2 = *(const v4f*)(ap + 16);
      v4f f3 = *(const v4f*)(ap + 20);
      __builtin_prefetch((const void*)(ap + 32), 0, 3);  // next K chunk
      v16bf a;
#pragma unroll
      for (int e = 0; e < 4; ++e) {
        a[e]      = (__bf16)f0[e];
        a[4 + e]  = (__bf16)f1[e];
        a[8 + e]  = (__bf16)f2[e];
        a[12 + e] = (__bf16)f3[e];
      }
#pragma unroll
      for (int j = 0; j < 4; ++j)
        acc[i][j] = __builtin_amdgcn_wmma_f32_16x16x32_bf16(
            false, a, false, bfrag[j], (short)0, acc[i][j], false, false);
    }
  }

  // --- epilogue: scale + bias, coalesced 64B row stores per half-wave ---
  const float scale = scale_p[0];
#pragma unroll
  for (int j = 0; j < 4; ++j) {
    const int n = n0 + j * 16 + r;
    const float b = bias[n];
#pragma unroll
    for (int i = 0; i < 4; ++i) {
      const int mb = m0 + i * 16 + hi * 8;
#pragma unroll
      for (int rr = 0; rr < 8; ++rr)
        out[(size_t)(mb + rr) * N_DIM + n] = acc[i][j][rr] * scale + b;
    }
  }
}

// ---------------------------------------------------------------------------
extern "C" void kernel_launch(void* const* d_in, const int* in_sizes, int n_in,
                              void* d_out, int out_size, void* d_ws,
                              size_t ws_size, hipStream_t stream) {
  (void)n_in; (void)out_size; (void)ws_size;
  const float* x    = (const float*)d_in[0];   // [M, K] fp32
  const float* w    = (const float*)d_in[1];   // [N, K] fp32
  const float* bias = (const float*)d_in[2];   // [N]    fp32
  float* out        = (float*)d_out;           // [M, N] fp32

  // workspace layout: [0,4KB) partials, [4KB,+4) scale, [8KB, 8KB+2MB) T bf16
  float*  part  = (float*)d_ws;
  float*  scale = part + 1024;
  __bf16* tq    = (__bf16*)((char*)d_ws + 8192);

  const int M = in_sizes[0] / K_DIM;           // 65536

  absmean_partial_kernel<<<dim3(1024), dim3(256), 0, stream>>>(w, part);
  absmean_final_kernel<<<dim3(1), dim3(256), 0, stream>>>(part, scale);
  quantize_kernel<<<dim3((N_DIM * K_DIM) / 256), dim3(256), 0, stream>>>(
      w, scale, tq);
  // N-tiles fastest in dispatch order -> same-M blocks share x lines in L2
  ternary_gemm_kernel<<<dim3(N_DIM / 256, M / 128), 256, 0, stream>>>(
      x, tq, bias, scale, out, M);
}
